// MultiHeadCrossAttention_64261300682818
// MI455X (gfx1250) — matrix operand order repaired and checked by reading, not compiled, not run on hardware
//
#include <hip/hip_runtime.h>
#include <hip/hip_bf16.h>
#include <math.h>

// MultiHeadCrossAttention on gfx1250 (MI455X), wave32 + WMMA f16->f32.
//
// Math notes vs reference:
//  * spatial bias sb is per-query, added uniformly over the softmax (key) axis
//    -> softmax-invariant -> omitted exactly.
//  * 1/sqrt(HEAD_DIM)=0.125 folded into Q at projection time.
//  * motion gate mg[k] multiplies attention AFTER softmax: flash recurrence
//    keeps ungated l = sum(e^{s-m}) and gated accumulator sum(e^{s-m}*mg*V).

#define N_TOK     2048
#define IN_DIM    256
#define OUT_DIM   512
#define NUM_HEADS 8
#define HEAD_DIM  64

typedef __attribute__((ext_vector_type(16))) _Float16 v16h;
typedef __attribute__((ext_vector_type(8)))  _Float16 v8h;
typedef __attribute__((ext_vector_type(8)))  float    v8f;

// Build a 16-half WMMA operand register from two contiguous 8-half (16B) loads.
__device__ __forceinline__ v16h ld16x2(const _Float16* p0, const _Float16* p1) {
  v8h a = *(const v8h*)p0;
  v8h b = *(const v8h*)p1;
  v16h r;
#pragma unroll
  for (int i = 0; i < 8; ++i) { r[i] = a[i]; r[i + 8] = b[i]; }
  return r;
}

// ---------------------------------------------------------------------------
// Kernel 1: f32 -> f16 conversion; x stays row-major, Wq/Wk/Wv transposed so
// WMMA B-operand columns (output-dim per lane) are contiguous along fan-in.
// ---------------------------------------------------------------------------
__global__ void k_convert(const float* __restrict__ x,
                          const float* __restrict__ Wq,
                          const float* __restrict__ Wk,
                          const float* __restrict__ Wv,
                          _Float16* __restrict__ x16,
                          _Float16* __restrict__ Wqt,
                          _Float16* __restrict__ Wkt,
                          _Float16* __restrict__ Wvt) {
  int idx = blockIdx.x * blockDim.x + threadIdx.x;
  const int NX = N_TOK * IN_DIM;    // 524288
  const int NW = OUT_DIM * IN_DIM;  // 131072
  if (idx < NX) {
    x16[idx] = (_Float16)x[idx];
  } else {
    int j = idx - NX;
    int mat = j / NW;
    int r = j - mat * NW;
    int o = r / IN_DIM;
    int i = r - o * IN_DIM;
    const float* W = (mat == 0) ? Wq : (mat == 1) ? Wk : Wv;
    _Float16* Wt   = (mat == 0) ? Wqt : (mat == 1) ? Wkt : Wvt;
    Wt[o * IN_DIM + i] = (_Float16)W[i * OUT_DIM + o];
  }
}

// ---------------------------------------------------------------------------
// Kernel 2: per-key motion gate mg[k] = sigmoid(relu([v,a]@Wmg1+b)@Wmg2+b)
// ---------------------------------------------------------------------------
__global__ void k_gate(const float* __restrict__ rel_vel,
                       const float* __restrict__ rel_angle,
                       const float* __restrict__ Wmg1, const float* __restrict__ bmg1,
                       const float* __restrict__ Wmg2, const float* __restrict__ bmg2,
                       float* __restrict__ mg) {
  int n = blockIdx.x * blockDim.x + threadIdx.x;
  if (n >= N_TOK) return;
  float v0 = rel_vel[n], v1 = rel_angle[n];
  float s = bmg2[0];
#pragma unroll 8
  for (int j = 0; j < HEAD_DIM; ++j) {
    float h = v0 * Wmg1[j] + v1 * Wmg1[HEAD_DIM + j] + bmg1[j];
    h = fmaxf(h, 0.0f);
    s += h * Wmg2[j];
  }
  mg[n] = 1.0f / (1.0f + __expf(-s));
}

// ---------------------------------------------------------------------------
// Kernel 3: QKV projection. One wave per 16(row)x16(col) output tile,
// K-loop 256/32 = 8 WMMAs. Q scaled by 0.125; V stored transposed (Vt[o][n]).
// ---------------------------------------------------------------------------
__global__ void k_qkv(const _Float16* __restrict__ x16,
                      const _Float16* __restrict__ Wqt,
                      const _Float16* __restrict__ Wkt,
                      const _Float16* __restrict__ Wvt,
                      const float* __restrict__ bq,
                      const float* __restrict__ bk,
                      const float* __restrict__ bv,
                      _Float16* __restrict__ Q16,
                      _Float16* __restrict__ K16,
                      _Float16* __restrict__ Vt16) {
  const int wave  = (blockIdx.x * blockDim.x + threadIdx.x) >> 5;
  const int lane  = threadIdx.x & 31;
  const int laneQ = lane & 15;
  const int hi    = lane >> 4;

  const int TILES = (N_TOK / 16) * (OUT_DIM / 16);  // 4096 tiles per matrix
  int which = wave / TILES;                         // 0=Q, 1=K, 2=V
  int t     = wave - which * TILES;
  int ntile = t / (OUT_DIM / 16);
  int otile = t - ntile * (OUT_DIM / 16);
  int n0 = ntile * 16, o0 = otile * 16;

  const _Float16* Wt = (which == 0) ? Wqt : (which == 1) ? Wkt : Wvt;
  const float*    b  = (which == 0) ? bq  : (which == 1) ? bk  : bv;

  const _Float16* arow = x16 + (n0 + laneQ) * IN_DIM;   // A: row M=laneQ
  const _Float16* brow = Wt  + (o0 + laneQ) * IN_DIM;   // B: column N=laneQ

  v8f acc = {};
#pragma unroll
  for (int kc = 0; kc < IN_DIM / 32; ++kc) {
    int kb = kc * 32;
    // A layout: lo lanes K0..15, hi lanes K16..31 (contiguous per row)
    v16h A = ld16x2(arow + kb + hi * 16, arow + kb + hi * 16 + 8);
    // B layout: lo lanes K0..15, hi lanes K16..31 (contiguous per column)
    v16h B = ld16x2(brow + kb + hi * 16, brow + kb + hi * 16 + 8);
    acc = __builtin_amdgcn_wmma_f32_16x16x32_f16(false, A, false, B,
                                                 (short)0, acc, false, false);
  }

  int   col   = o0 + laneQ;
  float bias  = b[col];
  float scale = (which == 0) ? 0.125f : 1.0f;  // fold 1/sqrt(64) into Q
  if (which < 2) {
    _Float16* dst = (which == 0) ? Q16 : K16;
#pragma unroll
    for (int r = 0; r < 8; ++r) {
      int row = n0 + r + hi * 8;  // C layout: M = r (+8 for hi lanes)
      dst[row * OUT_DIM + col] = (_Float16)((acc[r] + bias) * scale);
    }
  } else {
#pragma unroll
    for (int r = 0; r < 8; ++r) {
      int row = n0 + r + hi * 8;
      Vt16[col * N_TOK + row] = (_Float16)(acc[r] + bias);  // V transposed
    }
  }
}

// ---------------------------------------------------------------------------
// Kernel 4: flash attention. One wave owns (16 queries x 1 head), loops over
// 2048 keys in 32-key chunks. Scores computed transposed (S^T = K x Q^T) so
// the key-axis softmax reduction is per-lane regs + one shfl_xor(16).
// O accumulated as O^T = Vt x P. 8 WMMAs + 16 exps per chunk.
// ---------------------------------------------------------------------------
__global__ void k_attn(const _Float16* __restrict__ Q16,
                       const _Float16* __restrict__ K16,
                       const _Float16* __restrict__ Vt16,
                       const float* __restrict__ mg,
                       float* __restrict__ Obuf) {
  const int wave  = (blockIdx.x * blockDim.x + threadIdx.x) >> 5;
  const int lane  = threadIdx.x & 31;
  const int laneQ = lane & 15;
  const int hi    = lane >> 4;

  const int h     = wave & (NUM_HEADS - 1);
  const int qtile = wave >> 3;
  const int q0    = qtile * 16;

  // Fixed B operand: Q^T (head-dim x queries); column q = laneQ, contiguous d.
  const _Float16* qrow = Q16 + (q0 + laneQ) * OUT_DIM + h * HEAD_DIM;
  v16h BQ1 = ld16x2(qrow + hi * 16,      qrow + hi * 16 + 8);       // d 0..31
  v16h BQ2 = ld16x2(qrow + 32 + hi * 16, qrow + 32 + hi * 16 + 8);  // d 32..63

  v8f acc[4] = {v8f{}, v8f{}, v8f{}, v8f{}};  // O^T tiles, d-tiles 0..3
  float m = -1e30f;  // running max for query laneQ (shared across lane pair)
  float l = 0.0f;    // ungated running sum (this lane's key half)

  for (int kb = 0; kb < N_TOK / 32; ++kb) {
    const int k0 = kb * 32;

    // --- score tiles: S^T[key][q], key tile M = laneQ-row of K ---
    v8f ct[2];
#pragma unroll
    for (int tt = 0; tt < 2; ++tt) {
      const _Float16* krow = K16 + (k0 + tt * 16 + laneQ) * OUT_DIM + h * HEAD_DIM;
      v16h A1 = ld16x2(krow + hi * 16,      krow + hi * 16 + 8);  // d 0..31
      v16h A2 = ld16x2(krow + 32 + hi * 16, krow + 40 + hi * 16);  // d 32..63
      v8f c = {};
      c = __builtin_amdgcn_wmma_f32_16x16x32_f16(false, A1, false, BQ1,
                                                 (short)0, c, false, false);
      c = __builtin_amdgcn_wmma_f32_16x16x32_f16(false, A2, false, BQ2,
                                                 (short)0, c, false, false);
      ct[tt] = c;
    }

    // --- online softmax stats (keys live along regs + partner lane) ---
    float cmax = -1e30f;
#pragma unroll
    for (int r = 0; r < 8; ++r) {
      cmax = fmaxf(cmax, ct[0][r]);
      cmax = fmaxf(cmax, ct[1][r]);
    }
    cmax = fmaxf(cmax, __shfl_xor(cmax, 16, 32));  // unify lane pair (same q)
    const float mnew = fmaxf(m, cmax);
    const float f    = __expf(m - mnew);
    m = mnew;
    l *= f;
#pragma unroll
    for (int r = 0; r < 8; ++r) {
      acc[0][r] *= f; acc[1][r] *= f; acc[2][r] *= f; acc[3][r] *= f;
    }

    float p1[8], p2[8];
#pragma unroll
    for (int r = 0; r < 8; ++r) {
      float e1 = __expf(ct[0][r] - mnew);  // key k0 +      r + hi*8
      float e2 = __expf(ct[1][r] - mnew);  // key k0 + 16 + r + hi*8
      l += e1 + e2;                        // denominator is UNgated
      p1[r] = e1 * mg[k0 + r + hi * 8];        // gated numerator terms
      p2[r] = e2 * mg[k0 + 16 + r + hi * 8];
    }

    // --- rearrange P into B-operand layout (32 keys x 16 q) ---
    // lo lane: col q, keys k0+0..7 = own p1, keys k0+8..15 = partner p1
    // hi lane: col q, keys k0+16..23 = partner p2, keys k0+24..31 = own p2
    v16h BP;
#pragma unroll
    for (int r = 0; r < 8; ++r) {
      float s1 = __shfl_xor(p1[r], 16, 32);
      float s2 = __shfl_xor(p2[r], 16, 32);
      BP[r]     = (_Float16)(hi ? s2    : p1[r]);
      BP[r + 8] = (_Float16)(hi ? p2[r] : s1);
    }

    // --- O^T += Vt_tile x P ---
#pragma unroll
    for (int dt = 0; dt < 4; ++dt) {
      const _Float16* vrow = Vt16 + (h * HEAD_DIM + dt * 16 + laneQ) * N_TOK;
      v16h AV = ld16x2(vrow + k0 + hi * 16, vrow + k0 + hi * 16 + 8);
      acc[dt] = __builtin_amdgcn_wmma_f32_16x16x32_f16(false, AV, false, BP,
                                                       (short)0, acc[dt],
                                                       false, false);
    }
  }

  // combine lane-pair denominators (both halves share m) and write O
  const float lsum = l + __shfl_xor(l, 16, 32);
  const float rinv = 1.0f / lsum;
  float* orow = Obuf + (q0 + laneQ) * OUT_DIM + h * HEAD_DIM;
#pragma unroll
  for (int dt = 0; dt < 4; ++dt)
#pragma unroll
    for (int r = 0; r < 8; ++r)
      orow[dt * 16 + r + hi * 8] = acc[dt][r] * rinv;
}

// ---------------------------------------------------------------------------
// Kernel 5: out[q] = O[q,:] . Wo + bo   (one wave per row, shuffle-reduce)
// ---------------------------------------------------------------------------
__global__ void k_proj(const float* __restrict__ Obuf,
                       const float* __restrict__ Wo,
                       const float* __restrict__ bo,
                       float* __restrict__ out) {
  const int wave = threadIdx.x >> 5;
  const int lane = threadIdx.x & 31;
  const int q = blockIdx.x * 8 + wave;
  const float* row = Obuf + q * OUT_DIM;
  float s = 0.0f;
#pragma unroll
  for (int i = lane; i < OUT_DIM; i += 32) s += row[i] * Wo[i];
#pragma unroll
  for (int off = 16; off > 0; off >>= 1) s += __shfl_xor(s, off, 32);
  if (lane == 0) out[q] = s + bo[0];
}

// ---------------------------------------------------------------------------
extern "C" void kernel_launch(void* const* d_in, const int* in_sizes, int n_in,
                              void* d_out, int out_size, void* d_ws, size_t ws_size,
                              hipStream_t stream) {
  (void)in_sizes; (void)n_in; (void)out_size; (void)ws_size;

  const float* x         = (const float*)d_in[0];
  const float* rel_vel   = (const float*)d_in[2];
  const float* rel_angle = (const float*)d_in[3];
  const float* Wq = (const float*)d_in[4];
  const float* bq = (const float*)d_in[5];
  const float* Wk = (const float*)d_in[6];
  const float* bk = (const float*)d_in[7];
  const float* Wv = (const float*)d_in[8];
  const float* bv = (const float*)d_in[9];
  // d_in[10..13] (spatial bias MLP) intentionally unused: softmax-invariant.
  const float* Wmg1 = (const float*)d_in[14];
  const float* bmg1 = (const float*)d_in[15];
  const float* Wmg2 = (const float*)d_in[16];
  const float* bmg2 = (const float*)d_in[17];
  const float* Wo = (const float*)d_in[18];
  const float* bo = (const float*)d_in[19];
  float* out = (float*)d_out;

  // workspace carve-up (all offsets 16B-aligned)
  char* ws = (char*)d_ws;
  _Float16* x16  = (_Float16*)(ws + 0);                 // 2048*256   (1 MB)
  _Float16* Wqt  = (_Float16*)(ws + 1048576);           // 512*256
  _Float16* Wkt  = (_Float16*)(ws + 1310720);
  _Float16* Wvt  = (_Float16*)(ws + 1572864);
  _Float16* Q16  = (_Float16*)(ws + 1835008);           // 2048*512   (2 MB)
  _Float16* K16  = (_Float16*)(ws + 3932160);
  _Float16* Vt16 = (_Float16*)(ws + 6029312);           // 512*2048
  float*    mg   = (float*)   (ws + 8126464);           // 2048
  float*    Obuf = (float*)   (ws + 8134656);           // 2048*512 f32 (4 MB)
  // total = 12,328,960 bytes

  // 1) conversions: 524288 + 3*131072 = 917504 elems = 3584 blocks exactly
  k_convert<<<3584, 256, 0, stream>>>(x, Wq, Wk, Wv, x16, Wqt, Wkt, Wvt);

  // 2) motion gate
  k_gate<<<N_TOK / 256, 256, 0, stream>>>(rel_vel, rel_angle, Wmg1, bmg1,
                                          Wmg2, bmg2, mg);

  // 3) QKV projections: 3 * 4096 tile-waves / 8 waves-per-block = 1536 blocks
  k_qkv<<<1536, 256, 0, stream>>>(x16, Wqt, Wkt, Wvt, bq, bk, bv,
                                  Q16, K16, Vt16);

  // 4) flash attention: 128 qtiles * 8 heads = 1024 waves = 128 blocks
  k_attn<<<128, 256, 0, stream>>>(Q16, K16, Vt16, mg, Obuf);

  // 5) output projection: 8 rows per block
  k_proj<<<N_TOK / 8, 256, 0, stream>>>(Obuf, Wo, bo, out);
}
